// VCA_Masking_80479097193197
// MI455X (gfx1250) — matrix operand order repaired and checked by reading, not compiled, not run on hardware
//
#include <hip/hip_runtime.h>

// ---------------------------------------------------------------------------
// Problem constants (reference: B=16, T=1024, D=512)
// ---------------------------------------------------------------------------
#define BB 16
#define TT 1024
#define DD 512
#define NTD (BB * TT * DD)           // 8388608

typedef __attribute__((ext_vector_type(16))) __bf16 v16bf;
typedef __attribute__((ext_vector_type(8)))  __bf16 v8bf;
typedef __attribute__((ext_vector_type(8)))  float  v8f;

// Types for the gfx1250 async global->LDS DMA builtin (signature learned from
// the compiler diagnostic: (v4i AS1*, v4i AS3*, imm offset, imm cpol)).
typedef int v4i_vs __attribute__((vector_size(16)));
typedef __attribute__((address_space(1))) v4i_vs* as1_v4i_ptr;
typedef __attribute__((address_space(3))) v4i_vs* as3_v4i_ptr;

// Optional gfx1250 async global->LDS DMA (ASYNCcnt path), guarded so the
// kernel still builds (with the synchronous staging path) on toolchains
// that do not expose the builtin.
#if defined(__has_builtin)
#if __has_builtin(__builtin_amdgcn_global_load_async_to_lds_b128)
#define HAVE_ASYNC_LDS 1
#endif
#if __has_builtin(__builtin_amdgcn_s_wait_asynccnt)
#define HAVE_WAIT_ASYNC 1
#endif
#endif

// ---------------------------------------------------------------------------
// WMMA fragment helpers (CDNA5 ISA 7.12.2 layouts, wave32)
// A 16x32 bf16: lane L holds row M=L%16; K = klo..klo+7 and klo+16..klo+23,
//               klo = (L<16)?0:8  -> two 16B chunks per lane.
// B 32x16 bf16: lane L holds col N=L%16; K = klo..klo+7 and klo+16..klo+23,
//               klo = (L<16)?0:8  -> two 16B chunks per lane.
// C/D f32 16x16: VGPR r, lane L -> M = r + 8*(L>>4), N = L&15.
// ---------------------------------------------------------------------------
__device__ __forceinline__ v16bf load_a_frag(const __bf16* p, int pitch, int lane) {
  const int row = lane & 15;
  const int klo = (lane >> 4) * 8;
  const __bf16* r0 = p + (long)row * pitch + klo;
  v8bf lo = *(const v8bf*)r0;
  v8bf hv = *(const v8bf*)(r0 + 16);
  return __builtin_shufflevector(lo, hv, 0, 1, 2, 3, 4, 5, 6, 7,
                                         8, 9, 10, 11, 12, 13, 14, 15);
}

__device__ __forceinline__ v16bf load_b_frag(const __bf16* p, int pitch, int lane) {
  const int row = lane & 15;
  const int koff = (lane >> 4) * 8;
  const __bf16* r0 = p + (long)row * pitch + koff;
  v8bf lo = *(const v8bf*)r0;
  v8bf hv = *(const v8bf*)(r0 + 16);
  return __builtin_shufflevector(lo, hv, 0, 1, 2, 3, 4, 5, 6, 7,
                                         8, 9, 10, 11, 12, 13, 14, 15);
}

__device__ __forceinline__ v8f wmma_bf16(v16bf a, v16bf b, v8f c) {
  return __builtin_amdgcn_wmma_f32_16x16x32_bf16(
      /*neg_a=*/false, a, /*neg_b=*/false, b,
      /*c_mod=*/(short)0, c, /*reuse_a=*/false, /*reuse_b=*/false);
}

__device__ __forceinline__ v8f zero8() {
  v8f z = {0.f, 0.f, 0.f, 0.f, 0.f, 0.f, 0.f, 0.f};
  return z;
}

// ---------------------------------------------------------------------------
// Prep kernels
// ---------------------------------------------------------------------------
__global__ __launch_bounds__(256) void cvt_inputs_kernel(
    const float* __restrict__ aud, const float* __restrict__ vid,
    __bf16* __restrict__ audb, __bf16* __restrict__ vidb,
    __bf16* __restrict__ fusedb) {
  const long i = (long)blockIdx.x * 256 + threadIdx.x;   // grid sized exactly
  audb[i] = (__bf16)aud[i];
  const float vv = vid[i];
  vidb[i] = (__bf16)vv;
  const long row = i >> 9;                // /DD
  const int  c   = (int)(i & (DD - 1));
  fusedb[row * 1024 + 512 + c] = (__bf16)vv;   // second half of fused input
}

// dst[n*dstStride + dstColOff + k] = src[k*N + n]   (fp32 -> bf16, transpose)
__global__ __launch_bounds__(256) void transpose_cvt_kernel(
    const float* __restrict__ src, __bf16* __restrict__ dst,
    int K, int N, int dstStride, int dstColOff) {
  const long i = (long)blockIdx.x * 256 + threadIdx.x;
  if (i >= (long)K * N) return;
  const int k = (int)(i / N);
  const int n = (int)(i % N);
  dst[(long)n * dstStride + dstColOff + k] = (__bf16)src[i];
}

// zero rows t=-1 and t=T of the padded value buffer (SAME conv boundary)
__global__ __launch_bounds__(256) void zero_valp_edges_kernel(__bf16* __restrict__ valp) {
  const int i = blockIdx.x * 256 + threadIdx.x;          // B*2*D = 16384
  const int b = i / (2 * DD);
  const int rem = i % (2 * DD);
  const int edge = rem / DD;
  const int c = rem % DD;
  const long row = (long)b * (TT + 2) + (edge ? (TT + 1) : 0);
  valp[row * DD + c] = (__bf16)0.0f;
}

// ---------------------------------------------------------------------------
// Generic LDS-staged WMMA GEMM:  C[M,N] = A[M,K] * Bt[N,K]^T + bias
// Block 256 threads (8 waves), tile 128x128, K-step 64.
// Wave grid 4(m) x 2(n); each wave: 2x4 tiles of 16x16.
// M,N multiples of 128; K multiple of 64 (all true here).
// Row m of A starts at m*lda + (m/batchT)*batchPad (supports the padded
// conv1 input where batches are separated by 2 pad rows).
// Staging uses GLOBAL_LOAD_ASYNC_TO_LDS_B128 (ASYNCcnt) when the toolchain
// exposes the builtin, else classic global_load_b128 + ds_store_b128.
// Epilogue MODE: 0 bf16, 1 bf16 transposed-per-batch (v -> v^T[b][d][t]),
//               2 f32, 3 relu->bf16, 4 sigmoid-gate -> fused buffer.
// ---------------------------------------------------------------------------
template <int MODE>
__global__ __launch_bounds__(256) void gemm_bf16_kernel(
    const __bf16* __restrict__ A, int lda, int batchT, int batchPad,
    const __bf16* __restrict__ Bt, int K,
    const float* __restrict__ bias,
    void* __restrict__ Cout, int ldc,
    const float* __restrict__ aud_f32, const int* __restrict__ vlen) {
  __shared__ __align__(16) __bf16 As[128 * 72];
  __shared__ __align__(16) __bf16 Bs[128 * 72];

  const int tid  = threadIdx.x;
  const int lane = tid & 31;
  const int w    = tid >> 5;
  const int mw   = w & 3;     // 0..3
  const int nw   = w >> 2;    // 0..1
  const long m0  = (long)blockIdx.x * 128;
  const int  n0  = blockIdx.y * 128;

  v8f acc[2][4];
#pragma unroll
  for (int i = 0; i < 2; ++i)
#pragma unroll
    for (int j = 0; j < 4; ++j) acc[i][j] = zero8();

  for (int k0 = 0; k0 < K; k0 += 64) {
    __syncthreads();   // WAR: previous stage fully consumed
#pragma unroll
    for (int it = 0; it < 4; ++it) {
      const int v = it * 256 + tid;
      const int r = v >> 3;            // 0..127
      const int c = (v & 7) * 8;       // 0,8,...,56
      const long rowA = m0 + r;
      long offA = rowA * (long)lda + (long)k0 + c;
      if (batchPad) offA += (rowA / batchT) * (long)batchPad;
      const long offB = (long)(n0 + r) * K + k0 + c;
#if defined(HAVE_ASYNC_LDS)
      // DMA 16B straight into LDS, no VGPR round-trip (ASYNCcnt tracked).
      __builtin_amdgcn_global_load_async_to_lds_b128(
          (as1_v4i_ptr)&A[offA], (as3_v4i_ptr)&As[r * 72 + c], 0, 0);
      __builtin_amdgcn_global_load_async_to_lds_b128(
          (as1_v4i_ptr)&Bt[offB], (as3_v4i_ptr)&Bs[r * 72 + c], 0, 0);
#else
      *(v8bf*)&As[r * 72 + c] = *(const v8bf*)&A[offA];
      *(v8bf*)&Bs[r * 72 + c] = *(const v8bf*)&Bt[offB];
#endif
    }
#if defined(HAVE_ASYNC_LDS)
#if defined(HAVE_WAIT_ASYNC)
    __builtin_amdgcn_s_wait_asynccnt(0);
#else
    asm volatile("s_wait_asynccnt 0" ::: "memory");
#endif
#endif
    __syncthreads();   // RAW: all waves' stages visible
#pragma unroll
    for (int kk = 0; kk < 64; kk += 32) {
      v16bf a0 = load_a_frag(&As[(mw * 32 + 0) * 72 + kk], 72, lane);
      v16bf a1 = load_a_frag(&As[(mw * 32 + 16) * 72 + kk], 72, lane);
#pragma unroll
      for (int j = 0; j < 4; ++j) {
        v16bf bj = load_b_frag(&Bs[(nw * 64 + j * 16) * 72 + kk], 72, lane);
        acc[0][j] = wmma_bf16(a0, bj, acc[0][j]);
        acc[1][j] = wmma_bf16(a1, bj, acc[1][j]);
      }
    }
  }

  const int nc = lane & 15;
  const int hi = lane >> 4;
#pragma unroll
  for (int i = 0; i < 2; ++i) {
#pragma unroll
    for (int j = 0; j < 4; ++j) {
      const int n = n0 + nw * 64 + j * 16 + nc;
      const float bn = bias[n];
#pragma unroll
      for (int r = 0; r < 8; ++r) {
        const long m = m0 + mw * 32 + i * 16 + r + 8 * hi;
        const float x = acc[i][j][r] + bn;
        if constexpr (MODE == 0) {
          ((__bf16*)Cout)[m * ldc + n] = (__bf16)x;
        } else if constexpr (MODE == 1) {
          const long bidx = m >> 10;            // /TT
          const long t = m & (TT - 1);
          ((__bf16*)Cout)[(bidx * DD + n) * (long)TT + t] = (__bf16)x;
        } else if constexpr (MODE == 2) {
          ((float*)Cout)[m * ldc + n] = x;
        } else if constexpr (MODE == 3) {
          ((__bf16*)Cout)[m * ldc + n] = (__bf16)fmaxf(x, 0.0f);
        } else {  // MODE == 4: gate epilogue -> fused[:, :512]
          const long bidx = m >> 10;
          const long t = m & (TT - 1);
          const float g = (t < (long)vlen[bidx]) ? (1.0f / (1.0f + __expf(-x))) : 0.0f;
          const float a = aud_f32[m * DD + n];
          ((__bf16*)Cout)[m * 1024 + n] = (__bf16)(a * (1.0f + g));
        }
      }
    }
  }
}

// ---------------------------------------------------------------------------
// Flash-attention kernel. One wave (32 threads) per 16 queries of one batch.
// S = (Q K^T) * 1/sqrt(D), key mask, online softmax, acc += P V.
// Q fragments stay resident in VGPRs (16 frags = 128 VGPRs); accumulator is
// 32 d-tiles (512 dims) of 16x16 f32. P is reshaped C-layout -> A-layout
// through a tiny per-wave LDS buffer (in-wave DS ordering + s_wait_dscnt).
// Output written (bf16) into the padded value buffer interior.
// ---------------------------------------------------------------------------
__global__ __launch_bounds__(32, 1) void attn_kernel(
    const __bf16* __restrict__ qb, const __bf16* __restrict__ kb,
    const __bf16* __restrict__ vtb, const int* __restrict__ vlen,
    __bf16* __restrict__ valp) {
  __shared__ __align__(16) __bf16 Pbuf[16 * 40];   // 16 rows, pitch 40 bf16

  const int lane = threadIdx.x;
  const int blk = blockIdx.x;
  const int b = blk >> 6;                  // T/16 = 64 query tiles per batch
  const int m0 = (blk & 63) * 16;
  const int nc = lane & 15;
  const int hi = lane >> 4;

  // Resident Q fragments for this 16-query tile (K = 512 -> 16 steps of 32)
  const __bf16* qbase = qb + ((long)b * TT + m0) * DD;
  v16bf qf[16];
#pragma unroll
  for (int ks = 0; ks < 16; ++ks) qf[ks] = load_a_frag(qbase + ks * 32, DD, lane);

  v8f acc[32];
#pragma unroll
  for (int dt = 0; dt < 32; ++dt) acc[dt] = zero8();

  float mrow[8], lrow[8];
#pragma unroll
  for (int r = 0; r < 8; ++r) { mrow[r] = -1e30f; lrow[r] = 0.0f; }

  const long vl = (long)vlen[b];
  const float sc = 0.044194173824159220f;  // 1/sqrt(512)

  for (int kc = 0; kc < TT; kc += 32) {
    // Prefetch next chunk's key rows / V^T rows toward the WGP while the
    // current chunk's WMMAs run (global_prefetch_b8).
    if (kc + 32 < TT) {
      const __bf16* knext = kb + ((long)b * TT + kc + 32 + (lane & 31)) * DD;
      __builtin_prefetch(knext, 0, 1);
      const __bf16* vnext = vtb + (long)b * DD * TT + (long)(lane * 16) * TT + kc + 32;
      __builtin_prefetch(vnext, 0, 1);
    }

    // ---- S tiles: 16 queries x 32 keys --------------------------------
    v8f s0 = zero8(), s1 = zero8();
    const __bf16* kbase = kb + ((long)b * TT + kc) * DD;
#pragma unroll
    for (int ks = 0; ks < 16; ++ks) {
      v16bf b0 = load_b_frag(kbase + ks * 32, DD, lane);
      v16bf b1 = load_b_frag(kbase + 16 * DD + ks * 32, DD, lane);
      s0 = wmma_bf16(qf[ks], b0, s0);
      s1 = wmma_bf16(qf[ks], b1, s1);
    }

    // ---- scale, mask, online softmax ----------------------------------
    const bool v0 = (kc + nc) < vl;         // key col of tile 0 (lane-uniform per r)
    const bool v1 = (kc + 16 + nc) < vl;    // key col of tile 1
    float p0[8], p1[8], alpha[8];
#pragma unroll
    for (int r = 0; r < 8; ++r) {
      const float x0 = v0 ? s0[r] * sc : -1e30f;
      const float x1 = v1 ? s1[r] * sc : -1e30f;
      float tm = fmaxf(x0, x1);
#pragma unroll
      for (int msk = 1; msk < 16; msk <<= 1) tm = fmaxf(tm, __shfl_xor(tm, msk, 32));
      const float mn = fmaxf(mrow[r], tm);
      alpha[r] = __expf(mrow[r] - mn);
      mrow[r] = mn;
      p0[r] = __expf(x0 - mn);
      p1[r] = __expf(x1 - mn);
      float rs = p0[r] + p1[r];
#pragma unroll
      for (int msk = 1; msk < 16; msk <<= 1) rs += __shfl_xor(rs, msk, 32);
      lrow[r] = lrow[r] * alpha[r] + rs;
    }
#pragma unroll
    for (int dt = 0; dt < 32; ++dt)
#pragma unroll
      for (int r = 0; r < 8; ++r) acc[dt][r] *= alpha[r];

    // ---- C-layout P -> A-layout via LDS -------------------------------
#pragma unroll
    for (int r = 0; r < 8; ++r) {
      Pbuf[(r + 8 * hi) * 40 + nc] = (__bf16)p0[r];
      Pbuf[(r + 8 * hi) * 40 + 16 + nc] = (__bf16)p1[r];
    }
    asm volatile("s_wait_dscnt 0" ::: "memory");   // in-wave DS RAW fence
    v16bf pf = load_a_frag(&Pbuf[0], 40, lane);

    // ---- acc += P (16x32) x V (32 keys x 512 dims), V stored transposed
    const __bf16* vtbase = vtb + (long)b * DD * TT + kc;
#pragma unroll
    for (int dt = 0; dt < 32; ++dt) {
      v16bf bv = load_b_frag(vtbase + (long)(dt * 16) * TT, TT, lane);
      acc[dt] = wmma_bf16(pf, bv, acc[dt]);
    }
  }

  // ---- normalize and store into padded value buffer interior ----------
  float inv[8];
#pragma unroll
  for (int r = 0; r < 8; ++r) inv[r] = 1.0f / lrow[r];
#pragma unroll
  for (int dt = 0; dt < 32; ++dt)
#pragma unroll
    for (int r = 0; r < 8; ++r) {
      const long mg = m0 + r + 8 * hi;
      valp[((long)b * (TT + 2) + 1 + mg) * DD + dt * 16 + nc] =
          (__bf16)(acc[dt][r] * inv[r]);
    }
}

// ---------------------------------------------------------------------------
// Host-side orchestration
// ---------------------------------------------------------------------------
extern "C" void kernel_launch(void* const* d_in, const int* in_sizes, int n_in,
                              void* d_out, int out_size, void* d_ws, size_t ws_size,
                              hipStream_t stream) {
  (void)in_sizes; (void)n_in; (void)out_size; (void)ws_size;

  const float* aud  = (const float*)d_in[0];
  const float* vid  = (const float*)d_in[1];
  const int*   vlen = (const int*)d_in[2];
  const float* Wq = (const float*)d_in[3];  const float* bq = (const float*)d_in[4];
  const float* Wk = (const float*)d_in[5];  const float* bk = (const float*)d_in[6];
  const float* Wv = (const float*)d_in[7];  const float* bv = (const float*)d_in[8];
  const float* Wc1 = (const float*)d_in[9]; const float* bc1 = (const float*)d_in[10];
  const float* Wc2 = (const float*)d_in[11];const float* bc2 = (const float*)d_in[12];
  const float* Wf = (const float*)d_in[13]; const float* bf = (const float*)d_in[14];

  // ---- workspace carve-up (bf16 elements), 256B aligned chunks ----------
  char* ws = (char*)d_ws;
  auto alloc_bf16 = [&](size_t elems) -> __bf16* {
    char* p = ws;
    ws += ((elems * sizeof(__bf16)) + 255) & ~(size_t)255;
    return (__bf16*)p;
  };
  __bf16* audb   = alloc_bf16(NTD);
  __bf16* vidb   = alloc_bf16(NTD);
  __bf16* qb     = alloc_bf16(NTD);
  __bf16* kb     = alloc_bf16(NTD);
  __bf16* vtb    = alloc_bf16(NTD);                       // v^T [B][D][T]
  __bf16* valp   = alloc_bf16((size_t)BB * (TT + 2) * DD);// padded value
  __bf16* h1     = alloc_bf16(NTD);                       // relu(conv1)
  __bf16* fusedb = alloc_bf16((size_t)BB * TT * 1024);    // [enh | vid]
  __bf16* WqT  = alloc_bf16(512 * 512);
  __bf16* WkT  = alloc_bf16(512 * 512);
  __bf16* WvT  = alloc_bf16(512 * 512);
  __bf16* Wc1T = alloc_bf16(512 * 1536);                  // [o][dt*512+i]
  __bf16* Wc2T = alloc_bf16(512 * 512);
  __bf16* WfT  = alloc_bf16(512 * 1024);

  // ---- 1) input conversion (fp32 -> bf16, also fills fused[:,512:]) -----
  cvt_inputs_kernel<<<NTD / 256, 256, 0, stream>>>(aud, vid, audb, vidb, fusedb);

  // ---- 2) weight transpose+convert --------------------------------------
  const int g512 = (512 * 512) / 256;
  transpose_cvt_kernel<<<g512, 256, 0, stream>>>(Wq, WqT, 512, 512, 512, 0);
  transpose_cvt_kernel<<<g512, 256, 0, stream>>>(Wk, WkT, 512, 512, 512, 0);
  transpose_cvt_kernel<<<g512, 256, 0, stream>>>(Wv, WvT, 512, 512, 512, 0);
  for (int dt = 0; dt < 3; ++dt)
    transpose_cvt_kernel<<<g512, 256, 0, stream>>>(Wc1 + (size_t)dt * 512 * 512,
                                                   Wc1T, 512, 512, 1536, dt * 512);
  transpose_cvt_kernel<<<g512, 256, 0, stream>>>(Wc2, Wc2T, 512, 512, 512, 0);
  transpose_cvt_kernel<<<(1024 * 512) / 256, 256, 0, stream>>>(Wf, WfT, 1024, 512, 1024, 0);

  zero_valp_edges_kernel<<<(BB * 2 * DD) / 256, 256, 0, stream>>>(valp);

  // ---- 3) Q/K/V projections (WMMA GEMMs) --------------------------------
  const dim3 gemm_grid(BB * TT / 128, DD / 128);  // (128, 4)
  gemm_bf16_kernel<0><<<gemm_grid, 256, 0, stream>>>(
      audb, DD, TT, 0, WqT, DD, bq, qb, DD, nullptr, nullptr);
  gemm_bf16_kernel<0><<<gemm_grid, 256, 0, stream>>>(
      vidb, DD, TT, 0, WkT, DD, bk, kb, DD, nullptr, nullptr);
  gemm_bf16_kernel<1><<<gemm_grid, 256, 0, stream>>>(   // -> v^T [B][D][T]
      vidb, DD, TT, 0, WvT, DD, bv, vtb, 0, nullptr, nullptr);

  // ---- 4) masked softmax attention (flash-style, WMMA) ------------------
  attn_kernel<<<BB * (TT / 16), 32, 0, stream>>>(qb, kb, vtb, vlen, valp);

  // ---- 5) conv1 (k=3 SAME) as K=1536 GEMM over padded rows + ReLU -------
  gemm_bf16_kernel<3><<<gemm_grid, 256, 0, stream>>>(
      valp, DD, TT, /*batchPad=*/2 * DD, Wc1T, 1536, bc1, h1, DD, nullptr, nullptr);

  // ---- 6) conv2 (k=1) + sigmoid gate -> fused[:, :512] ------------------
  gemm_bf16_kernel<4><<<gemm_grid, 256, 0, stream>>>(
      h1, DD, TT, 0, Wc2T, DD, bc2, fusedb, 1024, aud, vlen);

  // ---- 7) final fused projection (K=1024) -> f32 output -----------------
  gemm_bf16_kernel<2><<<gemm_grid, 256, 0, stream>>>(
      fusedb, 1024, TT, 0, WfT, 1024, bf, d_out, DD, nullptr, nullptr);
}
